// CausalSelfAttention_8486855377166
// MI455X (gfx1250) — matrix operand order repaired and checked by reading, not compiled, not run on hardware
//
#include <hip/hip_runtime.h>
#include <hip/hip_bf16.h>

typedef __attribute__((ext_vector_type(16))) __bf16 v16bf;
typedef __attribute__((ext_vector_type(8)))  __bf16 v8bf;
typedef __attribute__((ext_vector_type(4)))  __bf16 v4bf;
typedef __attribute__((ext_vector_type(8)))  float  v8f;
typedef unsigned int v4u __attribute__((ext_vector_type(4)));
typedef int          v8i __attribute__((ext_vector_type(8)));
typedef int          v4i __attribute__((ext_vector_type(4)));

#define WMMA_BF16(a, b, c) \
  __builtin_amdgcn_wmma_f32_16x16x32_bf16(false, (a), false, (b), (short)0, (c), false, false)

constexpr int Bb  = 2;
constexpr int Tt  = 2048;
constexpr int Cc  = 1024;
constexpr int Hh  = 16;
constexpr int Dd  = 64;
constexpr int C2  = 2 * Cc;     // 2048
constexpr int C3  = 3 * Cc;     // 3072
constexpr int Mm  = Bb * Tt;    // 4096

__device__ inline v8f vzero8() {
  v8f z;
#pragma unroll
  for (int i = 0; i < 8; ++i) z[i] = 0.0f;
  return z;
}

// A-fragment (16x32 bf16, MxK): lane<16 -> M=lane, K=0..15; lane>=16 -> M=lane-16, K=16..31
__device__ inline v16bf load_a_frag(const __bf16* __restrict__ base, int ld, int row0, int k0, int lane) {
  const int r  = lane & 15;
  const int hi = lane >> 4;
  const __bf16* p = base + (size_t)(row0 + r) * ld + k0 + hi * 16;
  v8bf lo = *(const v8bf*)(p);
  v8bf hh = *(const v8bf*)(p + 8);
  v16bf a;
#pragma unroll
  for (int i = 0; i < 8; ++i) { a[i] = lo[i]; a[8 + i] = hh[i]; }
  return a;
}

// B-fragment (32x16 bf16, KxN): lane holds column N=lane&15, 16 consecutive K values per lane half.
__device__ inline v16bf load_b_frag_lds(const __bf16* ldsB, int n0t, int lane) {
  const int n  = lane & 15;
  const int hi = lane >> 4;
  return *(const v16bf*)(ldsB + (n0t + n) * 32 + hi * 16);
}

// ---- Issue staging of a 64(N-rows) x 32(K) bf16 weight tile into LDS (no wait here).
// Preferred: Tensor Data Mover DMA issued by wave 0. Fallback: cooperative copy by all 256 threads.
__device__ inline void stage_issue(const __bf16* __restrict__ gsrc, int ld, int rows_total,
                                   int n0, int k0, __bf16* ldsB) {
#if __has_builtin(__builtin_amdgcn_tensor_load_to_lds) && __has_builtin(__builtin_amdgcn_s_wait_tensorcnt)
  if (threadIdx.x < 32) {                    // wave 0 issues the TDM op (EXEC ignored, per-wave issue)
    const unsigned long long ga = (unsigned long long)(uintptr_t)(gsrc + (size_t)n0 * ld + k0);
    const unsigned int ldsoff   = (unsigned int)(uintptr_t)ldsB;  // LDS byte offset (low 32 bits)
    v4u g0;
    g0[0] = 1u;                                                   // count=1, user mode
    g0[1] = ldsoff;                                               // lds_addr
    g0[2] = (unsigned int)ga;                                     // global_addr[31:0]
    g0[3] = (unsigned int)((ga >> 32) & 0x01FFFFFFu) | (2u << 30);// global_addr[56:32] | type=2
    v8i g1;
    g1[0] = 1 << 16;                                              // data_size=1 (2 bytes)
    g1[1] = (int)(((unsigned)ld & 0xFFFFu) << 16);                // tensor_dim0[15:0]
    g1[2] = (int)((((unsigned)ld >> 16) & 0xFFFFu) |
                  (((unsigned)rows_total & 0xFFFFu) << 16));      // tensor_dim0[31:16] | tensor_dim1[15:0]
    g1[3] = (int)((((unsigned)rows_total >> 16) & 0xFFFFu) |
                  (32u << 16));                                   // tensor_dim1[31:16] | tile_dim0=32
    g1[4] = 64;                                                   // tile_dim1=64, tile_dim2=0
    g1[5] = ld;                                                   // tensor_dim0_stride[31:0]
    g1[6] = 0;                                                    // stride0[47:32] | dim1_stride[15:0]
    g1[7] = 0;
    v4i z4;
#pragma unroll
    for (int i = 0; i < 4; ++i) z4[i] = 0;
    v8i z8;
#pragma unroll
    for (int i = 0; i < 8; ++i) z8[i] = 0;
    __builtin_amdgcn_tensor_load_to_lds(g0, g1, z4, z4, z8, 0);
  }
#else
  const int idx = threadIdx.x;               // 256 threads x 16B = 4KB tile
  v8bf tmp = *(const v8bf*)(gsrc + (size_t)(n0 + (idx >> 2)) * ld + k0 + (idx & 3) * 8);
  *(v8bf*)&ldsB[(idx >> 2) * 32 + (idx & 3) * 8] = tmp;
#endif
}

// ---- Wait for outstanding tile DMA and make it visible block-wide.
__device__ inline void stage_wait() {
#if __has_builtin(__builtin_amdgcn_tensor_load_to_lds) && __has_builtin(__builtin_amdgcn_s_wait_tensorcnt)
  __builtin_amdgcn_s_wait_tensorcnt(0);
#endif
  __syncthreads();
}

// ---- One K-step of the 32x64 wave tile: A loads + 8 WMMAs from an LDS-resident B tile.
__device__ inline void gemm_step(const __bf16* __restrict__ abase, int m0, int k0, int lane,
                                 const __bf16* ldsB, v8f acc[2][4]) {
  if (k0 + 256 < Cc)
    __builtin_prefetch(abase + (size_t)(m0 + (lane & 15)) * Cc + k0 + 256, 0, 0);
  v16bf a0 = load_a_frag(abase, Cc, m0, k0, lane);
  v16bf a1 = load_a_frag(abase, Cc, m0 + 16, k0, lane);
#pragma unroll
  for (int t = 0; t < 4; ++t) {
    v16bf bfr = load_b_frag_lds(ldsB, t * 16, lane);
    acc[0][t] = WMMA_BF16(a0, bfr, acc[0][t]);
    acc[1][t] = WMMA_BF16(a1, bfr, acc[1][t]);
  }
}

// ---------------- f32 -> bf16 conversion ----------------
__global__ __launch_bounds__(256) void cvt_f32_bf16(const float* __restrict__ src,
                                                    __bf16* __restrict__ dst, int n4) {
  int i = blockIdx.x * blockDim.x + threadIdx.x;
  if (i < n4) {
    float4 f = ((const float4*)src)[i];
    v4bf o;
    o[0] = (__bf16)f.x; o[1] = (__bf16)f.y; o[2] = (__bf16)f.z; o[3] = (__bf16)f.w;
    ((v4bf*)dst)[i] = o;
  }
}

// ---------------- QKV GEMM: [4096,1024] x [3072,1024]^T + bias ----------------
// Each wave: 32(M) x 64(N). Weight tiles double-buffered in LDS via TDM, DMA overlapped with WMMA.
// Writes q,k into qk [B*T, 2C] (bf16) and v transposed into vt [B,H,D,T] (bf16).
__global__ __launch_bounds__(256) void gemm_qkv(const __bf16* __restrict__ xb,
                                                const __bf16* __restrict__ wab,
                                                const float* __restrict__ bias,
                                                __bf16* __restrict__ qk,
                                                __bf16* __restrict__ vt) {
  __shared__ __bf16 ldsB[2][64 * 32];
  const int lane = threadIdx.x & 31;
  const int wave = threadIdx.x >> 5;
  const int m0 = blockIdx.x * 256 + wave * 32;   // 16 blocks in x
  const int n0 = blockIdx.y * 64;                // 48 blocks in y

  v8f acc[2][4];
#pragma unroll
  for (int mi = 0; mi < 2; ++mi)
#pragma unroll
    for (int t = 0; t < 4; ++t) acc[mi][t] = vzero8();

  stage_issue(wab, Cc, C3, n0, 0, ldsB[0]);
  for (int k0 = 0; k0 < Cc; k0 += 64) {
    stage_wait();                                        // tile k0 in buf0 ready
    stage_issue(wab, Cc, C3, n0, k0 + 32, ldsB[1]);      // overlap DMA with compute
    gemm_step(xb, m0, k0, lane, ldsB[0], acc);
    stage_wait();                                        // tile k0+32 in buf1 ready
    if (k0 + 64 < Cc)
      stage_issue(wab, Cc, C3, n0, k0 + 64, ldsB[0]);
    gemm_step(xb, m0, k0 + 32, lane, ldsB[1], acc);
  }

  const int hi = lane >> 4;
  const int nn = lane & 15;
#pragma unroll
  for (int mi = 0; mi < 2; ++mi) {
#pragma unroll
    for (int t = 0; t < 4; ++t) {
      const int o = n0 + t * 16 + nn;
      const float bv = bias[o];
#pragma unroll
      for (int vg = 0; vg < 8; ++vg) {
        const int row = m0 + mi * 16 + vg + 8 * hi;
        const __bf16 val = (__bf16)(acc[mi][t][vg] + bv);
        if (o < C2) {
          qk[(size_t)row * C2 + o] = val;               // q at o<C, k at C<=o<2C
        } else {
          const int d  = o - C2;
          const int h  = d >> 6;
          const int dd = d & 63;
          const int tq = row & (Tt - 1);
          const int bbx = row >> 11;                    // row / T
          vt[(size_t)(((bbx * Hh + h) * Dd) + dd) * Tt + tq] = val;
        }
      }
    }
  }
}

// ---------------- Flash attention (causal) ----------------
// One wave = one 16-row q tile of one (b,h). 32-key steps, online softmax.
__global__ __launch_bounds__(256) void attn_fwd(const __bf16* __restrict__ qk,
                                                const __bf16* __restrict__ vt,
                                                __bf16* __restrict__ yb) {
  __shared__ __bf16 plds[8][16][32];   // per-wave P staging (C-layout -> A-layout)

  const int lane = threadIdx.x & 31;
  const int wave = threadIdx.x >> 5;
  const int nqc  = Tt / 128;                 // q chunks per (b,h)
  const int bh   = blockIdx.x / nqc;         // 0..31
  const int qc   = blockIdx.x % nqc;
  const int bbx  = bh >> 4;
  const int h    = bh & 15;
  const int q0   = qc * 128 + wave * 16;

  const __bf16* qbase = qk + (size_t)(bbx * Tt) * C2 + h * Dd;
  const __bf16* kbase = qbase + Cc;
  const __bf16* vbase = vt + (size_t)bh * Dd * Tt;

  const int hi = lane >> 4;
  const int ln = lane & 15;

  // q A-fragments: d 0..31 and d 32..63
  v16bf qf[2];
#pragma unroll
  for (int f = 0; f < 2; ++f) {
    const __bf16* p = qbase + (size_t)(q0 + ln) * C2 + f * 32 + hi * 16;
    v8bf lo = *(const v8bf*)(p);
    v8bf hh = *(const v8bf*)(p + 8);
#pragma unroll
    for (int i = 0; i < 8; ++i) { qf[f][i] = lo[i]; qf[f][8 + i] = hh[i]; }
  }

  float mrow[8], lrow[8];
  v8f o_acc[4];
#pragma unroll
  for (int vg = 0; vg < 8; ++vg) { mrow[vg] = -3.0e38f; lrow[vg] = 0.0f; }
#pragma unroll
  for (int dt = 0; dt < 4; ++dt) o_acc[dt] = vzero8();

  const float scale = 0.125f;                // 1/sqrt(64)
  const int tq_base = q0 + 8 * hi;           // row for vgpr vg = tq_base + vg
  const int kend    = q0 + 16;               // causal: keys <= q0+15

  for (int t0 = 0; t0 < kend; t0 += 32) {
    // ---- S = q @ k^T : two 16x16 tiles over 32 keys ----
    v8f s[2];
#pragma unroll
    for (int st = 0; st < 2; ++st) {
      v8f z = vzero8();
      const __bf16* krow = kbase + (size_t)(t0 + st * 16 + ln) * C2;
      v16bf kb0 = *(const v16bf*)(krow + hi * 16);
      v16bf kb1 = *(const v16bf*)(krow + 32 + hi * 16);
      z = WMMA_BF16(qf[0], kb0, z);
      z = WMMA_BF16(qf[1], kb1, z);
      s[st] = z;
    }

    // ---- causal mask + scale ----
#pragma unroll
    for (int st = 0; st < 2; ++st) {
      const int tk = t0 + st * 16 + ln;
#pragma unroll
      for (int vg = 0; vg < 8; ++vg) {
        const float sv = s[st][vg] * scale;
        s[st][vg] = (tk <= tq_base + vg) ? sv : -3.0e38f;
      }
    }

    // ---- online softmax: row max over 32 keys (reduce across 16-lane halves) ----
    float mnew[8];
#pragma unroll
    for (int vg = 0; vg < 8; ++vg) {
      float v = fmaxf(s[0][vg], s[1][vg]);
      v = fmaxf(v, __shfl_xor(v, 1, 32));
      v = fmaxf(v, __shfl_xor(v, 2, 32));
      v = fmaxf(v, __shfl_xor(v, 4, 32));
      v = fmaxf(v, __shfl_xor(v, 8, 32));
      mnew[vg] = fmaxf(mrow[vg], v);
    }
#pragma unroll
    for (int vg = 0; vg < 8; ++vg) {
      const float alpha = __expf(mrow[vg] - mnew[vg]);
      const float p0 = __expf(s[0][vg] - mnew[vg]);
      const float p1 = __expf(s[1][vg] - mnew[vg]);
      s[0][vg] = p0;
      s[1][vg] = p1;
      float ps = p0 + p1;
      ps += __shfl_xor(ps, 1, 32);
      ps += __shfl_xor(ps, 2, 32);
      ps += __shfl_xor(ps, 4, 32);
      ps += __shfl_xor(ps, 8, 32);
      lrow[vg] = lrow[vg] * alpha + ps;
      mrow[vg] = mnew[vg];
#pragma unroll
      for (int dt = 0; dt < 4; ++dt) o_acc[dt][vg] *= alpha;
    }

    // ---- P: C-layout -> LDS -> A-layout fragment ----
#pragma unroll
    for (int st = 0; st < 2; ++st)
#pragma unroll
      for (int vg = 0; vg < 8; ++vg)
        plds[wave][vg + 8 * hi][st * 16 + ln] = (__bf16)s[st][vg];

    v16bf pf;
    {
      const __bf16* p = &plds[wave][ln][hi * 16];
      v8bf lo = *(const v8bf*)(p);
      v8bf hh = *(const v8bf*)(p + 8);
#pragma unroll
      for (int i = 0; i < 8; ++i) { pf[i] = lo[i]; pf[8 + i] = hh[i]; }
    }

    // ---- O += P @ V (V pre-transposed: rows are d, K=t contiguous) ----
#pragma unroll
    for (int dt = 0; dt < 4; ++dt) {
      v16bf vb = *(const v16bf*)(vbase + (size_t)(dt * 16 + ln) * Tt + t0 + hi * 16);
      o_acc[dt] = WMMA_BF16(pf, vb, o_acc[dt]);
    }
  }

  // ---- normalize and store y (bf16, [B*T, C]) ----
#pragma unroll
  for (int vg = 0; vg < 8; ++vg) {
    const float inv = 1.0f / lrow[vg];
    const int tq = q0 + vg + 8 * hi;
    const size_t rowoff = (size_t)(bbx * Tt + tq) * Cc + h * Dd;
#pragma unroll
    for (int dt = 0; dt < 4; ++dt)
      yb[rowoff + dt * 16 + ln] = (__bf16)(o_acc[dt][vg] * inv);
  }
}

// ---------------- Output projection: [4096,1024] x [1024,1024]^T + bias -> f32 ----------------
__global__ __launch_bounds__(256) void gemm_proj(const __bf16* __restrict__ yb,
                                                 const __bf16* __restrict__ wpb,
                                                 const float* __restrict__ bias,
                                                 float* __restrict__ out) {
  __shared__ __bf16 ldsB[2][64 * 32];
  const int lane = threadIdx.x & 31;
  const int wave = threadIdx.x >> 5;
  const int m0 = blockIdx.x * 256 + wave * 32;   // 16 blocks
  const int n0 = blockIdx.y * 64;                // 16 blocks

  v8f acc[2][4];
#pragma unroll
  for (int mi = 0; mi < 2; ++mi)
#pragma unroll
    for (int t = 0; t < 4; ++t) acc[mi][t] = vzero8();

  stage_issue(wpb, Cc, Cc, n0, 0, ldsB[0]);
  for (int k0 = 0; k0 < Cc; k0 += 64) {
    stage_wait();
    stage_issue(wpb, Cc, Cc, n0, k0 + 32, ldsB[1]);
    gemm_step(yb, m0, k0, lane, ldsB[0], acc);
    stage_wait();
    if (k0 + 64 < Cc)
      stage_issue(wpb, Cc, Cc, n0, k0 + 64, ldsB[0]);
    gemm_step(yb, m0, k0 + 32, lane, ldsB[1], acc);
  }

  const int hi = lane >> 4;
  const int nn = lane & 15;
#pragma unroll
  for (int mi = 0; mi < 2; ++mi) {
#pragma unroll
    for (int t = 0; t < 4; ++t) {
      const int o = n0 + t * 16 + nn;
      const float bv = bias[o];
#pragma unroll
      for (int vg = 0; vg < 8; ++vg) {
        const int row = m0 + mi * 16 + vg + 8 * hi;
        out[(size_t)row * Cc + o] = acc[mi][t][vg] + bv;
      }
    }
  }
}

extern "C" void kernel_launch(void* const* d_in, const int* in_sizes, int n_in,
                              void* d_out, int out_size, void* d_ws, size_t ws_size,
                              hipStream_t stream) {
  const float* x      = (const float*)d_in[0];
  const float* w_attn = (const float*)d_in[1];
  const float* b_attn = (const float*)d_in[2];
  const float* w_proj = (const float*)d_in[3];
  const float* b_proj = (const float*)d_in[4];
  float* out = (float*)d_out;

  char* ws = (char*)d_ws;
  size_t off = 0;
  auto wsalloc = [&](size_t bytes) -> char* {
    char* p = ws + off;
    off = (off + bytes + 255) & ~(size_t)255;
    return p;
  };

  __bf16* xb  = (__bf16*)wsalloc((size_t)Mm * Cc * 2);           // x bf16
  __bf16* wab = (__bf16*)wsalloc((size_t)C3 * Cc * 2);           // w_attn bf16
  __bf16* wpb = (__bf16*)wsalloc((size_t)Cc * Cc * 2);           // w_proj bf16
  __bf16* qk  = (__bf16*)wsalloc((size_t)Mm * C2 * 2);           // q,k  [B*T, 2C]
  __bf16* vt  = (__bf16*)wsalloc((size_t)Bb * Hh * Dd * Tt * 2); // v^T [B,H,D,T]
  __bf16* yb  = (__bf16*)wsalloc((size_t)Mm * Cc * 2);           // attention output bf16

  {
    int n4 = Mm * Cc / 4;
    cvt_f32_bf16<<<(n4 + 255) / 256, 256, 0, stream>>>(x, xb, n4);
  }
  {
    int n4 = C3 * Cc / 4;
    cvt_f32_bf16<<<(n4 + 255) / 256, 256, 0, stream>>>(w_attn, wab, n4);
  }
  {
    int n4 = Cc * Cc / 4;
    cvt_f32_bf16<<<(n4 + 255) / 256, 256, 0, stream>>>(w_proj, wpb, n4);
  }

  gemm_qkv<<<dim3(Mm / 256, C3 / 64), 256, 0, stream>>>(xb, wab, b_attn, qk, vt);
  attn_fwd<<<Bb * Hh * (Tt / 128), 256, 0, stream>>>(qk, vt, yb);
  gemm_proj<<<dim3(Mm / 256, Cc / 64), 256, 0, stream>>>(yb, wpb, b_proj, out);
}
